// PaiNNPropertyPredictor_38843684225096
// MI455X (gfx1250) — matrix operand order repaired and checked by reading, not compiled, not run on hardware
//
#include <hip/hip_runtime.h>
#include <math.h>

#define NNODES  10000
#define NEDGES  160000
#define DDIM    128
#define EDIM    20
#define NMOL    500
#define RROUNDS 3
#define GBLOCKS 3
#define CUTOFF  5.0f

typedef __attribute__((ext_vector_type(16))) _Float16 v16h;
typedef __attribute__((ext_vector_type(8)))  _Float16 v8h;
typedef __attribute__((ext_vector_type(8)))  float    v8f;

// ---------------------------------------------------------------------------
// WMMA GEMM: Out[M,DOUT] = act(A[M,K] @ W[K,DOUT] + bias)
// Grid: x = groups of 8 M-tiles (one wave32 per 16x16 tile), y = N-tile.
// All 8 waves of a block share one 16-column W tile, staged in LDS as f16,
// TRANSPOSED ([n][k], row pad 8 halves) so each lane's 16 B-operands per
// WMMA step are two runs of 8 consecutive halves, 16 apart -> two ds_load_b128, no addr chains.
// A tile (16-bit A 16x32, ISA 7.12.2): lane = M%16; lanes<16 hold K 0-7 /
// 16-23, lanes>=16 hold K 8-15 / 24-31 (2 K per VGPR) -> float2 global loads.
// B tile (32x16): element h of v16h maps to K = kk + hi*8 + h (h<8) / kk + 16 + hi*8 + (h-8) (h>=8), N = lane%16.
// C/D f32: VGPR r -> row M=r (lanes 0-15) / M=r+8 (lanes 16-31), N=lane%16.
// ---------------------------------------------------------------------------
template <int K, int DOUT>
__global__ __launch_bounds__(256) void gemm_wmma(const float* __restrict__ A,
                                                 const float* __restrict__ W,
                                                 const float* __restrict__ bias,
                                                 float* __restrict__ Out,
                                                 int M, int act) {
  constexpr int KP = K + 8;                 // padded LDS row stride (halves)
  __shared__ _Float16 sWT[16 * KP];
  const int nBase = blockIdx.y << 4;
  for (int e = threadIdx.x; e < K * 16; e += blockDim.x) {
    const int k = e >> 4, n = e & 15;       // coalesced 64B rows of W
    sWT[n * KP + k] = (_Float16)W[(size_t)k * DOUT + nBase + n];
  }
  __syncthreads();

  const int tm = blockIdx.x * (blockDim.x >> 5) + (threadIdx.x >> 5);
  if (tm >= (M >> 4)) return;               // wave-uniform guard
  const int lane  = threadIdx.x & 31;
  const int hi    = lane >> 4;              // 0: lanes 0-15, 1: 16-31
  const int mBase = tm << 4;
  const int rowA  = mBase + (lane & 15);
  const int colB  = nBase + (lane & 15);
  const _Float16* wrow = sWT + (lane & 15) * KP + (hi << 3);

  v8f c = {};
#pragma unroll
  for (int kk = 0; kk < K; kk += 32) {
    v16h a;
#pragma unroll
    for (int j = 0; j < 8; ++j) {
      const int kbase = ((j < 4) ? 0 : 16) + (hi << 3) + ((j & 3) << 1);
      const float2 f2 = *(const float2*)(A + (size_t)rowA * K + kk + kbase);
      a[2 * j]     = (_Float16)f2.x;
      a[2 * j + 1] = (_Float16)f2.y;
    }
    const v8h b0 = *(const v8h*)(wrow + kk);        // ds_load_b128
    const v8h b1 = *(const v8h*)(wrow + kk + 16);   // ds_load_b128
    v16h b;
#pragma unroll
    for (int i = 0; i < 8; ++i) { b[i] = b0[i]; b[8 + i] = b1[i]; }
    c = __builtin_amdgcn_wmma_f32_16x16x32_f16(false, a, false, b,
                                               (short)0, c, false, false);
  }
#pragma unroll
  for (int r = 0; r < 8; ++r) {
    const int row = mBase + r + (hi << 3);
    float val = c[r];
    if (bias) val += bias[colB];
    if (act)  val = val / (1.0f + __expf(-val));   // SiLU
    Out[(size_t)row * DOUT + colB] = val;
  }
}

// ---------------------------------------------------------------------------
// Edge geometry: unit vectors, sinc RBF, cosine cutoff.
// ---------------------------------------------------------------------------
__global__ void edge_geom(const float* __restrict__ pos, const int* __restrict__ ei,
                          float* __restrict__ unit, float* __restrict__ rbf,
                          float* __restrict__ fcut, int E) {
  int e = blockIdx.x * blockDim.x + threadIdx.x;
  if (e >= E) return;
  const int rn = ei[e], cn = ei[E + e];
  const float rx = pos[3 * cn]     - pos[3 * rn];
  const float ry = pos[3 * cn + 1] - pos[3 * rn + 1];
  const float rz = pos[3 * cn + 2] - pos[3 * rn + 2];
  float d = sqrtf(rx * rx + ry * ry + rz * rz);
  d = fmaxf(d, 1e-6f);
  const float inv = 1.0f / d;
  unit[3 * e]     = rx * inv;
  unit[3 * e + 1] = ry * inv;
  unit[3 * e + 2] = rz * inv;
  const float pid = 3.14159265358979f * d / CUTOFF;
#pragma unroll
  for (int k = 0; k < EDIM; ++k)
    rbf[(size_t)e * EDIM + k] = __sinf((float)(k + 1) * pid) * inv;
  fcut[e] = (d < CUTOFF) ? 0.5f * (__cosf(pid) + 1.0f) : 0.0f;
}

// ---------------------------------------------------------------------------
// Node embedding: s[n,d] = h[n,:5] @ emb_w + emb_b   (K=5: plain VALU)
// ---------------------------------------------------------------------------
__global__ void embed(const float* __restrict__ h, const float* __restrict__ w,
                      const float* __restrict__ b, float* __restrict__ s, int n) {
  int i = blockIdx.x * blockDim.x + threadIdx.x;
  if (i >= n) return;
  const int node = i >> 7, d = i & 127;
  float acc = b[d];
#pragma unroll
  for (int k = 0; k < 5; ++k) acc = fmaf(h[node * 5 + k], w[k * DDIM + d], acc);
  s[i] = acc;
}

__global__ void zero_f(float* __restrict__ p, size_t n) {
  size_t i = (size_t)blockIdx.x * blockDim.x + threadIdx.x;
  if (i < n) p[i] = 0.0f;
}

__global__ void add_inplace(float* __restrict__ dst, const float* __restrict__ src,
                            size_t n) {
  size_t i = (size_t)blockIdx.x * blockDim.x + threadIdx.x;
  if (i < n) dst[i] += src[i];
}

// ---------------------------------------------------------------------------
// Fused edge message: filt = (rbf @ Wf + bf) * fcut (Wf in LDS),
// x = phi[col] * filt; scatter xs -> s, (xvv*v[col] + xvw*unit) -> dv.
// One wave32 per edge, 4 D-channels per lane.
// ---------------------------------------------------------------------------
__global__ __launch_bounds__(256) void edge_message(
    const float* __restrict__ phi, const float* __restrict__ rbf,
    const float* __restrict__ fcut, const float* __restrict__ unit,
    const int* __restrict__ rown, const int* __restrict__ coln,
    const float* __restrict__ fw, const float* __restrict__ fb,
    const float* __restrict__ vx, const float* __restrict__ vy,
    const float* __restrict__ vz,
    float* __restrict__ s, float* __restrict__ dvx, float* __restrict__ dvy,
    float* __restrict__ dvz, int E) {
  __shared__ float sW[EDIM * 384];
  __shared__ float sB[384];
  for (int i = threadIdx.x; i < EDIM * 384; i += blockDim.x) sW[i] = fw[i];
  for (int i = threadIdx.x; i < 384; i += blockDim.x) sB[i] = fb[i];
  __syncthreads();

  const int lane        = threadIdx.x & 31;
  const int wavesPerBlk = blockDim.x >> 5;
  const int nWaves      = gridDim.x * wavesPerBlk;
  for (int e = blockIdx.x * wavesPerBlk + (threadIdx.x >> 5); e < E; e += nWaves) {
    const int rn = rown[e], cn = coln[e];
    const float fc = fcut[e];
    const float ux = unit[3 * e], uy = unit[3 * e + 1], uz = unit[3 * e + 2];
    float rb[EDIM];
#pragma unroll
    for (int k = 0; k < EDIM; ++k) rb[k] = rbf[(size_t)e * EDIM + k];
    const float* prow  = phi + (size_t)cn * 384;
    const size_t cb    = (size_t)cn * DDIM;
    const size_t rbase = (size_t)rn * DDIM;
#pragma unroll
    for (int q = 0; q < 4; ++q) {
      const int d = (lane << 2) + q;
      float f0 = sB[d], f1 = sB[d + 128], f2 = sB[d + 256];
#pragma unroll
      for (int k = 0; k < EDIM; ++k) {
        const float* wr = sW + k * 384;
        f0 = fmaf(rb[k], wr[d], f0);
        f1 = fmaf(rb[k], wr[d + 128], f1);
        f2 = fmaf(rb[k], wr[d + 256], f2);
      }
      const float xs  = prow[d]       * (f0 * fc);
      const float xvv = prow[d + 128] * (f1 * fc);
      const float xvw = prow[d + 256] * (f2 * fc);
      atomicAdd(&s[rbase + d], xs);
      atomicAdd(&dvx[rbase + d], fmaf(xvv, vx[cb + d], xvw * ux));
      atomicAdd(&dvy[rbase + d], fmaf(xvv, vy[cb + d], xvw * uy));
      atomicAdd(&dvz[rbase + d], fmaf(xvv, vz[cb + d], xvw * uz));
    }
  }
}

// cat[n,:128] = s ; cat[n,128:] = ||Vv||  (safe norm)
__global__ void cat_norm(const float* __restrict__ s, const float* __restrict__ ax,
                         const float* __restrict__ ay, const float* __restrict__ az,
                         float* __restrict__ cat, int n) {
  int i = blockIdx.x * blockDim.x + threadIdx.x;
  if (i >= n) return;
  const int node = i >> 7, d = i & 127;
  cat[(size_t)node * 256 + d] = s[i];
  const float x = ax[i], y = ay[i], z = az[i];
  cat[(size_t)node * 256 + 128 + d] =
      sqrtf(fmaf(x, x, fmaf(y, y, fmaf(z, z, 1e-8f))));
}

// PaiNN update epilogue: s += a_ss + a_sv*<Uv,Vv> ; v += a_vv*Uv
__global__ void update_final(const float* __restrict__ a,
                             const float* __restrict__ ux, const float* __restrict__ uy,
                             const float* __restrict__ uz,
                             const float* __restrict__ wx, const float* __restrict__ wy,
                             const float* __restrict__ wz,
                             float* __restrict__ s, float* __restrict__ vx,
                             float* __restrict__ vy, float* __restrict__ vz, int n) {
  int i = blockIdx.x * blockDim.x + threadIdx.x;
  if (i >= n) return;
  const int node = i >> 7, d = i & 127;
  const size_t ab = (size_t)node * 384 + d;
  const float dot = ux[i] * wx[i] + uy[i] * wy[i] + uz[i] * wz[i];
  s[i] += a[ab] + a[ab + 128] * dot;
  const float g = a[ab + 256];
  vx[i] += g * ux[i];
  vy[i] += g * uy[i];
  vz[i] += g * uz[i];
}

// GEB epilogue: s = a[:, :128] ; v = a[:,128:] * v1
__global__ void geb_final(const float* __restrict__ a,
                          const float* __restrict__ ux, const float* __restrict__ uy,
                          const float* __restrict__ uz,
                          float* __restrict__ s, float* __restrict__ vx,
                          float* __restrict__ vy, float* __restrict__ vz, int n) {
  int i = blockIdx.x * blockDim.x + threadIdx.x;
  if (i >= n) return;
  const int node = i >> 7, d = i & 127;
  const size_t ab = (size_t)node * 256 + d;
  s[i] = a[ab];
  const float g = a[ab + 128];
  vx[i] = g * ux[i];
  vy[i] = g * uy[i];
  vz[i] = g * uz[i];
}

// Readout tail: per-node dot(t[n,:64], w2)+b2, scatter-add per molecule.
__global__ void readout(const float* __restrict__ t, const float* __restrict__ w2,
                        const float* __restrict__ b2, const int* __restrict__ batch,
                        float* __restrict__ out, int n) {
  int node = blockIdx.x * blockDim.x + threadIdx.x;
  if (node >= n) return;
  float acc = b2[0];
#pragma unroll
  for (int k = 0; k < 64; ++k) acc = fmaf(t[(size_t)node * 64 + k], w2[k], acc);
  atomicAdd(&out[batch[node]], acc);
}

// ---------------------------------------------------------------------------
static inline void launch_gemm(int K, const float* A, const float* W,
                               const float* bias, float* Out, int M, int Dout,
                               int act, hipStream_t st) {
  dim3 grid((unsigned)(((M >> 4) + 7) >> 3), (unsigned)(Dout >> 4));
  if (K == 128) {
    if (Dout == 128)
      gemm_wmma<128, 128><<<grid, 256, 0, st>>>(A, W, bias, Out, M, act);
    else if (Dout == 384)
      gemm_wmma<128, 384><<<grid, 256, 0, st>>>(A, W, bias, Out, M, act);
    else if (Dout == 256)
      gemm_wmma<128, 256><<<grid, 256, 0, st>>>(A, W, bias, Out, M, act);
    else
      gemm_wmma<128, 64><<<grid, 256, 0, st>>>(A, W, bias, Out, M, act);
  } else {
    gemm_wmma<256, 128><<<grid, 256, 0, st>>>(A, W, bias, Out, M, act);
  }
}

extern "C" void kernel_launch(void* const* d_in, const int* in_sizes, int n_in,
                              void* d_out, int out_size, void* d_ws, size_t ws_size,
                              hipStream_t stream) {
  (void)in_sizes; (void)n_in; (void)out_size; (void)ws_size;
  const float* h     = (const float*)d_in[0];
  const float* pos   = (const float*)d_in[1];
  const int*   ei    = (const int*)d_in[2];
  const int*   batch = (const int*)d_in[3];
  const float* emb_w = (const float*)d_in[4];
  const float* emb_b = (const float*)d_in[5];
  const float* mp_w1 = (const float*)d_in[6];
  const float* mp_b1 = (const float*)d_in[7];
  const float* mp_w2 = (const float*)d_in[8];
  const float* mp_b2 = (const float*)d_in[9];
  const float* mf_w  = (const float*)d_in[10];
  const float* mf_b  = (const float*)d_in[11];
  const float* u_U   = (const float*)d_in[12];
  const float* u_V   = (const float*)d_in[13];
  const float* u_w1  = (const float*)d_in[14];
  const float* u_b1  = (const float*)d_in[15];
  const float* u_w2  = (const float*)d_in[16];
  const float* u_b2  = (const float*)d_in[17];
  const float* g_wv1 = (const float*)d_in[18];
  const float* g_wv2 = (const float*)d_in[19];
  const float* g_w1  = (const float*)d_in[20];
  const float* g_b1  = (const float*)d_in[21];
  const float* g_w2  = (const float*)d_in[22];
  const float* g_b2  = (const float*)d_in[23];
  const float* ro_w1 = (const float*)d_in[24];
  const float* ro_b1 = (const float*)d_in[25];
  const float* ro_w2 = (const float*)d_in[26];
  const float* ro_b2 = (const float*)d_in[27];
  float* out = (float*)d_out;

  float* ws = (float*)d_ws;
  const size_t ND = (size_t)NNODES * DDIM;
  float* s    = ws;
  float* vx   = ws + 1 * ND;
  float* vy   = ws + 2 * ND;
  float* vz   = ws + 3 * ND;
  float* dvx  = ws + 4 * ND;            // dv[xyz] contiguous (3*ND)
  float* dvy  = ws + 5 * ND;
  float* dvz  = ws + 6 * ND;
  float* Uvx  = ws + 7 * ND;
  float* Uvy  = ws + 8 * ND;
  float* Uvz  = ws + 9 * ND;
  float* Vvx  = ws + 10 * ND;
  float* Vvy  = ws + 11 * ND;
  float* Vvz  = ws + 12 * ND;
  float* t1   = ws + 13 * ND;           // [N,128] intermediate (also [N,64])
  float* cat  = ws + 14 * ND;           // [N,256]
  float* buf3 = ws + 16 * ND;           // [N,384]: phi, then update/GEB "a"
  float* unit = ws + 19 * ND;           // [E,3]
  float* rbfb = unit + 3 * (size_t)NEDGES;    // [E,20]
  float* fcut = rbfb + (size_t)EDIM * NEDGES; // [E]

  const int T = 256;
  const int gND  = (int)((ND + T - 1) / T);
  const int g3ND = (int)((3 * ND + T - 1) / T);

  edge_geom<<<(NEDGES + T - 1) / T, T, 0, stream>>>(pos, ei, unit, rbfb, fcut, NEDGES);
  embed<<<gND, T, 0, stream>>>(h, emb_w, emb_b, s, (int)ND);
  zero_f<<<g3ND, T, 0, stream>>>(vx, 3 * ND);   // vx,vy,vz contiguous

  for (int r = 0; r < RROUNDS; ++r) {
    // phi = silu(s@W1+b1)@W2+b2
    launch_gemm(128, s, mp_w1 + (size_t)r * 128 * 128, mp_b1 + (size_t)r * 128,
                t1, NNODES, 128, 1, stream);
    launch_gemm(128, t1, mp_w2 + (size_t)r * 128 * 384, mp_b2 + (size_t)r * 384,
                buf3, NNODES, 384, 0, stream);
    // message scatter
    zero_f<<<g3ND, T, 0, stream>>>(dvx, 3 * ND);
    edge_message<<<2048, 256, 0, stream>>>(buf3, rbfb, fcut, unit, ei, ei + NEDGES,
                                           mf_w + (size_t)r * EDIM * 384,
                                           mf_b + (size_t)r * 384,
                                           vx, vy, vz, s, dvx, dvy, dvz, NEDGES);
    add_inplace<<<g3ND, T, 0, stream>>>(vx, dvx, 3 * ND);
    // update: Uv = v@U, Vv = v@V (per component)
    const float* U = u_U + (size_t)r * 128 * 128;
    const float* V = u_V + (size_t)r * 128 * 128;
    launch_gemm(128, vx, U, nullptr, Uvx, NNODES, 128, 0, stream);
    launch_gemm(128, vy, U, nullptr, Uvy, NNODES, 128, 0, stream);
    launch_gemm(128, vz, U, nullptr, Uvz, NNODES, 128, 0, stream);
    launch_gemm(128, vx, V, nullptr, Vvx, NNODES, 128, 0, stream);
    launch_gemm(128, vy, V, nullptr, Vvy, NNODES, 128, 0, stream);
    launch_gemm(128, vz, V, nullptr, Vvz, NNODES, 128, 0, stream);
    cat_norm<<<gND, T, 0, stream>>>(s, Vvx, Vvy, Vvz, cat, (int)ND);
    launch_gemm(256, cat, u_w1 + (size_t)r * 256 * 128, u_b1 + (size_t)r * 128,
                t1, NNODES, 128, 1, stream);
    launch_gemm(128, t1, u_w2 + (size_t)r * 128 * 384, u_b2 + (size_t)r * 384,
                buf3, NNODES, 384, 0, stream);
    update_final<<<gND, T, 0, stream>>>(buf3, Uvx, Uvy, Uvz, Vvx, Vvy, Vvz,
                                        s, vx, vy, vz, (int)ND);
  }

  for (int g = 0; g < GBLOCKS; ++g) {
    const float* W1 = g_wv1 + (size_t)g * 128 * 128;
    const float* W2 = g_wv2 + (size_t)g * 128 * 128;
    launch_gemm(128, vx, W1, nullptr, Uvx, NNODES, 128, 0, stream);   // v1
    launch_gemm(128, vy, W1, nullptr, Uvy, NNODES, 128, 0, stream);
    launch_gemm(128, vz, W1, nullptr, Uvz, NNODES, 128, 0, stream);
    launch_gemm(128, vx, W2, nullptr, Vvx, NNODES, 128, 0, stream);   // v2
    launch_gemm(128, vy, W2, nullptr, Vvy, NNODES, 128, 0, stream);
    launch_gemm(128, vz, W2, nullptr, Vvz, NNODES, 128, 0, stream);
    cat_norm<<<gND, T, 0, stream>>>(s, Vvx, Vvy, Vvz, cat, (int)ND);
    launch_gemm(256, cat, g_w1 + (size_t)g * 256 * 128, g_b1 + (size_t)g * 128,
                t1, NNODES, 128, 1, stream);
    launch_gemm(128, t1, g_w2 + (size_t)g * 128 * 256, g_b2 + (size_t)g * 256,
                buf3, NNODES, 256, 0, stream);
    geb_final<<<gND, T, 0, stream>>>(buf3, Uvx, Uvy, Uvz, s, vx, vy, vz, (int)ND);
  }

  // readout
  launch_gemm(128, s, ro_w1, ro_b1, t1, NNODES, 64, 1, stream);
  zero_f<<<(NMOL + T - 1) / T, T, 0, stream>>>(out, (size_t)NMOL);
  readout<<<(NNODES + T - 1) / T, T, 0, stream>>>(t1, ro_w2, ro_b2, batch, out, NNODES);
}
